// SelfAttention_71665824301384
// MI455X (gfx1250) — matrix operand order repaired and checked by reading, not compiled, not run on hardware
//
#include <hip/hip_runtime.h>
#include <hip/hip_bf16.h>
#include <cstdint>

typedef __attribute__((ext_vector_type(16))) __bf16 v16bf;
typedef __attribute__((ext_vector_type(8)))  __bf16 v8bf;
typedef __attribute__((ext_vector_type(8)))  float  v8f;

static constexpr int Bt = 8, Nt = 1024, Ht = 8;
static constexpr int HID_MV = 8, HID_S = 16, MVD = 32;
static constexpr int ROWS = Bt * Nt;               // 8192 tokens
static constexpr int FEAT = 64;                    // attention feature dim (8*6 + 16)
static constexpr int VDIM = HID_MV * MVD + HID_S;  // 272 packed value cols

// ---------------------------------------------------------------- WMMA helpers

__device__ __forceinline__ v8f wmma_bf16(v16bf a, v16bf b, v8f c) {
  // D = A(16x32) * B(32x16) + C, f32 accumulate
  return __builtin_amdgcn_wmma_f32_16x16x32_bf16(false, a, false, b, (short)0, c,
                                                 false, false);
}

// A fragment: 16(M) x 32(K) bf16 from row-major X (ld elems).
// ISA 7.12.2: lanes 0-15 row M=lane hold K={0..7,16..23}; lanes 16-31 hold K={8..15,24..31}.
__device__ __forceinline__ v16bf load_A_frag(const __bf16* Xrow, int ld, int kBase, int lane) {
  int m  = lane & 15;
  int hi = (lane >> 4) & 1;
  const __bf16* p = Xrow + (size_t)m * ld + kBase + hi * 8;
  v8bf lo = *(const v8bf*)(p);
  v8bf hh = *(const v8bf*)(p + 16);
  v16bf a;
#pragma unroll
  for (int e = 0; e < 8; ++e) { a[e] = lo[e]; a[e + 8] = hh[e]; }
  return a;
}

// B fragment: 32(K) x 16(N) where B[k][n] = W[n][k]; W row-major (n, k).
// lanes 0-15: col n, K={0..7,16..23}; lanes 16-31: col n, K={8..15,24..31}.
__device__ __forceinline__ v16bf load_B_nk(const __bf16* W, int ld, int nBase, int kBase, int lane) {
  int n  = nBase + (lane & 15);
  int k0 = kBase + ((lane >> 4) & 1) * 8;
  const __bf16* p = W + (size_t)n * ld + k0;
  v8bf lo = *(const v8bf*)(p);
  v8bf hh = *(const v8bf*)(p + 16);
  v16bf b;
#pragma unroll
  for (int e = 0; e < 8; ++e) { b[e] = lo[e]; b[e + 8] = hh[e]; }
  return b;
}

// ---------------------------------------------------------------- DPP row reductions
// Butterfly over a 16-lane DPP row: quad_perm(xor1), quad_perm(xor2),
// row_half_mirror, row_mirror. wave32's two rows reduce independently.

template <int CTRL>
__device__ __forceinline__ float dpp_xfer(float v) {
  int x = __float_as_int(v);
  int y = __builtin_amdgcn_update_dpp(x, x, CTRL, 0xF, 0xF, true);
  return __int_as_float(y);
}

__device__ __forceinline__ float row16_max(float v) {
  v = fmaxf(v, dpp_xfer<0xB1>(v));   // quad_perm [1,0,3,2]  (xor 1)
  v = fmaxf(v, dpp_xfer<0x4E>(v));   // quad_perm [2,3,0,1]  (xor 2)
  v = fmaxf(v, dpp_xfer<0x141>(v));  // row_half_mirror
  v = fmaxf(v, dpp_xfer<0x140>(v));  // row_mirror
  return v;
}

__device__ __forceinline__ float row16_sum(float v) {
  v += dpp_xfer<0xB1>(v);
  v += dpp_xfer<0x4E>(v);
  v += dpp_xfer<0x141>(v);
  v += dpp_xfer<0x140>(v);
  return v;
}

// ---------------------------------------------------------------- generic GEMM
// C[m,n] = sum_k X[m,k] * W[n,k];  X:(M,K) bf16, W:(Nc,K) bf16, C:(M,Nc) OutT.
// One wave computes a 16 x (16*NT) strip; k-loop is software-pipelined.
template <typename OutT, int NT>
__global__ __launch_bounds__(32) void gemm_bf16_wmma(const __bf16* __restrict__ X,
                                                     const __bf16* __restrict__ W,
                                                     OutT* __restrict__ C,
                                                     int M, int Nc, int K) {
  const int lane    = threadIdx.x;
  const int nStrips = Nc / (16 * NT);
  const int mBase   = (blockIdx.x / nStrips) << 4;
  const int nBase   = (blockIdx.x % nStrips) * (16 * NT);
  const v8f vzero = {0.f, 0.f, 0.f, 0.f, 0.f, 0.f, 0.f, 0.f};
  v8f acc[NT];
#pragma unroll
  for (int t = 0; t < NT; ++t) acc[t] = vzero;

  const __bf16* Xrow = X + (size_t)mBase * K;
  v16bf a = load_A_frag(Xrow, K, 0, lane);
  v16bf bcur[NT];
#pragma unroll
  for (int t = 0; t < NT; ++t) bcur[t] = load_B_nk(W, K, nBase + 16 * t, 0, lane);

  for (int k = 0; k < K; k += 32) {
    const int kn = (k + 32 < K) ? k + 32 : k;  // clamped prefetch index
    v16bf aN = load_A_frag(Xrow, K, kn, lane);
    v16bf bN[NT];
#pragma unroll
    for (int t = 0; t < NT; ++t) bN[t] = load_B_nk(W, K, nBase + 16 * t, kn, lane);
#pragma unroll
    for (int t = 0; t < NT; ++t) acc[t] = wmma_bf16(a, bcur[t], acc[t]);
    a = aN;
#pragma unroll
    for (int t = 0; t < NT; ++t) bcur[t] = bN[t];
  }

  const int mOff = mBase + ((lane >> 4) & 1) * 8;
#pragma unroll
  for (int t = 0; t < NT; ++t) {
    const int n = nBase + 16 * t + (lane & 15);
#pragma unroll
    for (int r = 0; r < 8; ++r)
      C[(size_t)(mOff + r) * Nc + n] = (OutT)acc[t][r];
  }
}

template <typename OutT, int NT>
static void launch_gemm(const __bf16* X, const __bf16* W, OutT* C, int M, int Nc, int K,
                        hipStream_t s) {
  gemm_bf16_wmma<OutT, NT><<<dim3((M / 16) * (Nc / (16 * NT))), dim3(32), 0, s>>>(X, W, C, M, Nc, K);
}

// ---------------------------------------------------------------- small prep kernels

__global__ void cvt_bf16(const float* __restrict__ src, __bf16* __restrict__ dst, long long n) {
  long long i = (long long)blockIdx.x * blockDim.x + threadIdx.x;
  if (i < n) dst[i] = (__bf16)src[i];
}

// W_eff[o*32+x][i*32+y] = sum_k w[o,i,k] * basis_pin[k,x,y]
__global__ void build_weff(const float* __restrict__ w, const float* __restrict__ bp,
                           __bf16* __restrict__ Weff, int Ochan, int Ichan) {
  const int cols = Ichan * 32;
  long long total = (long long)Ochan * 32 * cols;
  long long idx = (long long)blockIdx.x * blockDim.x + threadIdx.x;
  if (idx >= total) return;
  int col  = (int)(idx % cols);
  int rowi = (int)(idx / cols);
  int o = rowi >> 5, x = rowi & 31;
  int i = col >> 5,  y = col & 31;
  float s = 0.f;
#pragma unroll
  for (int k = 0; k < 9; ++k)
    s += w[(o * Ichan + i) * 9 + k] * bp[(k * 32 + x) * 32 + y];
  Weff[(size_t)rowi * cols + col] = (__bf16)s;
}

// Query features: Qf[(b*H+h)*N+n][c*6+f] = scale * qvec^T Bq[:,:,f] qvec; [48+p] = scale*q_s
__global__ __launch_bounds__(32) void qf_pack(const __bf16* __restrict__ Qmv,
                                              const __bf16* __restrict__ Qs,
                                              const float* __restrict__ basis_q,
                                              __bf16* __restrict__ Qf, float scale) {
  const int row  = blockIdx.x;        // (b*H + h)*N + n
  const int lane = threadIdx.x;
  const int n = row & (Nt - 1);
  const int h = (row >> 10) & 7;
  const int b = row >> 13;
  const size_t xrow = (size_t)(b * Nt + n);
  if (lane < 8) {
    const int c = lane;
    const __bf16* q = Qmv + xrow * 2048 + (h * HID_MV + c) * MVD + 1;
    float v[5];
#pragma unroll
    for (int x = 0; x < 5; ++x) v[x] = (float)q[x];
#pragma unroll
    for (int f = 0; f < 6; ++f) {
      float s = 0.f;
#pragma unroll
      for (int x = 0; x < 5; ++x)
#pragma unroll
        for (int y = 0; y < 5; ++y)
          s += v[x] * basis_q[(x * 5 + y) * 6 + f] * v[y];
      Qf[(size_t)row * FEAT + c * 6 + f] = (__bf16)(s * scale);
    }
  } else if (lane < 24) {
    const int p = lane - 8;
    Qf[(size_t)row * FEAT + 48 + p] = (__bf16)(scale * (float)Qs[xrow * 128 + h * HID_S + p]);
  }
}

// Key features (shared across heads, no scale)
__global__ __launch_bounds__(32) void kf_pack(const __bf16* __restrict__ Kmv,
                                              const __bf16* __restrict__ Ks,
                                              const float* __restrict__ basis_k,
                                              __bf16* __restrict__ Kfo) {
  const int row  = blockIdx.x;        // b*N + n
  const int lane = threadIdx.x;
  if (lane < 8) {
    const int c = lane;
    const __bf16* k = Kmv + (size_t)row * 256 + c * MVD + 1;
    float v[5];
#pragma unroll
    for (int x = 0; x < 5; ++x) v[x] = (float)k[x];
#pragma unroll
    for (int f = 0; f < 6; ++f) {
      float s = 0.f;
#pragma unroll
      for (int x = 0; x < 5; ++x)
#pragma unroll
        for (int y = 0; y < 5; ++y)
          s += v[x] * basis_k[(x * 5 + y) * 6 + f] * v[y];
      Kfo[(size_t)row * FEAT + c * 6 + f] = (__bf16)s;
    }
  } else if (lane < 24) {
    const int p = lane - 8;
    Kfo[(size_t)row * FEAT + 48 + p] = Ks[(size_t)row * 16 + p];
  }
}

// Transposed packed values: Vt[(b*272 + col)*N + key]; col 0..255 = v_mv, 256.. = v_s.
// Makes the attention B-fragment loads contiguous (2 x b128 per lane).
__global__ void vt_pack(const __bf16* __restrict__ Vmv, const __bf16* __restrict__ Vs,
                        __bf16* __restrict__ Vt) {
  const long long TOT = (long long)Bt * VDIM * Nt;
  long long i = (long long)blockIdx.x * blockDim.x + threadIdx.x;
  if (i >= TOT) return;
  const int n   = (int)(i % Nt);
  const int rem = (int)(i / Nt);
  const int col = rem % VDIM;
  const int b   = rem / VDIM;
  const size_t row = (size_t)(b * Nt + n);
  Vt[i] = (col < 256) ? Vmv[row * 256 + col] : Vs[row * 16 + (col - 256)];
}

// ---------------------------------------------------------------- fused flash attention
// One wave handles 16 queries of one (b,h); loops over 32-key blocks; keeps 17 f32
// D-tiles (272 value cols) live across the K loop with online softmax.
// K fragments double-buffered, V fragments double-buffered, next V block prefetched.
__global__ __launch_bounds__(32) void attn_fused(const __bf16* __restrict__ Qf,
                                                 const __bf16* __restrict__ Kf,
                                                 const __bf16* __restrict__ Vt,
                                                 __bf16* __restrict__ Hmv,
                                                 __bf16* __restrict__ Hs,
                                                 __bf16* __restrict__ Hmv0) {
  __shared__ __align__(16) __bf16 Plds[16 * 32];
  const int lane = threadIdx.x;
  const int blk = blockIdx.x;          // b*512 + h*64 + qt
  const int qt = blk & 63;
  const int h  = (blk >> 6) & 7;
  const int b  = blk >> 9;
  constexpr int NBLK = Nt / 32;

  const __bf16* Qbase = Qf + ((size_t)((b * Ht + h) * Nt) + qt * 16) * FEAT;
  const v16bf qa0 = load_A_frag(Qbase, FEAT, 0, lane);
  const v16bf qa1 = load_A_frag(Qbase, FEAT, 32, lane);

  const __bf16* Kfb = Kf + (size_t)(b * Nt) * FEAT;
  const __bf16* Vtb = Vt + (size_t)(b * VDIM) * Nt;

  v8f acc[17];
  const v8f vzero = {0.f, 0.f, 0.f, 0.f, 0.f, 0.f, 0.f, 0.f};
#pragma unroll
  for (int t = 0; t < 17; ++t) acc[t] = vzero;
  float mrun[8], lrun[8];
#pragma unroll
  for (int r = 0; r < 8; ++r) { mrun[r] = -3.0e38f; lrun[r] = 0.f; }

  // Prime K fragment double buffer (block 0)
  v16bf k00 = load_B_nk(Kfb, FEAT, 0, 0, lane);
  v16bf k01 = load_B_nk(Kfb, FEAT, 0, 32, lane);
  v16bf k10 = load_B_nk(Kfb, FEAT, 16, 0, lane);
  v16bf k11 = load_B_nk(Kfb, FEAT, 16, 32, lane);

  for (int kb = 0; kb < NBLK; ++kb) {
    const int kk = kb * 32;
    v8f S0 = vzero, S1 = vzero;
    S0 = wmma_bf16(qa0, k00, S0);
    S0 = wmma_bf16(qa1, k01, S0);
    S1 = wmma_bf16(qa0, k10, S1);
    S1 = wmma_bf16(qa1, k11, S1);

    // Prefetch next key block's K fragments (clamped) while softmax runs.
    const int kbn = (kb + 1 < NBLK) ? kb + 1 : kb;
    const __bf16* Knxt = Kfb + (size_t)(kbn * 32) * FEAT;
    v16bf n00 = load_B_nk(Knxt, FEAT, 0, 0, lane);
    v16bf n01 = load_B_nk(Knxt, FEAT, 0, 32, lane);
    v16bf n10 = load_B_nk(Knxt, FEAT, 16, 0, lane);
    v16bf n11 = load_B_nk(Knxt, FEAT, 16, 32, lane);

    // L2 prefetch of next block's Vt region (272 rows x 32 keys).
    {
      const __bf16* Vn = Vtb + (size_t)(kbn * 32) + ((lane >> 4) << 4);
#pragma unroll
      for (int t = 0; t < 17; ++t)
        __builtin_prefetch(Vn + (size_t)(t * 16 + (lane & 15)) * Nt, 0, 1);
    }

    // Online softmax: D-tile lane L holds (row = r + 8*(L>=16), col = L&15);
    // row reductions use DPP16 butterflies within each 16-lane row.
#pragma unroll
    for (int r = 0; r < 8; ++r) {
      const float v = row16_max(fmaxf(S0[r], S1[r]));
      const float mnew  = fmaxf(mrun[r], v);
      const float alpha = __expf(mrun[r] - mnew);
      const float p0 = __expf(S0[r] - mnew);
      const float p1 = __expf(S1[r] - mnew);
      S0[r] = p0;
      S1[r] = p1;
      const float rs = row16_sum(p0 + p1);
      lrun[r] = lrun[r] * alpha + rs;
      mrun[r] = mnew;
#pragma unroll
      for (int t = 0; t < 17; ++t) acc[t][r] *= alpha;
    }

    // Re-shape P from D layout to A layout through LDS (16 x 32, row-major)
    {
      const int rowOff = ((lane >> 4) & 1) * 8;
      const int col = lane & 15;
#pragma unroll
      for (int r = 0; r < 8; ++r) {
        Plds[(rowOff + r) * 32 + col]      = (__bf16)S0[r];
        Plds[(rowOff + r) * 32 + 16 + col] = (__bf16)S1[r];
      }
    }
    __syncthreads();
    const v16bf pA = load_A_frag(Plds, 32, 0, lane);
    __syncthreads();

    // P @ V over 17 column tiles, V fragments double-buffered.
    v16bf vcur = load_B_nk(Vtb, Nt, 0, kk, lane);
#pragma unroll
    for (int t = 0; t < 17; ++t) {
      v16bf vnxt = (t < 16) ? load_B_nk(Vtb, Nt, (t + 1) * 16, kk, lane) : vcur;
      acc[t] = wmma_bf16(pA, vcur, acc[t]);
      vcur = vnxt;
    }

    k00 = n00; k01 = n01; k10 = n10; k11 = n11;
  }

  // Epilogue: normalize and scatter into packed H layouts for the output GEMMs.
  const int mOff = ((lane >> 4) & 1) * 8;
  const int nl = lane & 15;
#pragma unroll
  for (int t = 0; t < 17; ++t) {
    const int col = t * 16 + nl;
#pragma unroll
    for (int r = 0; r < 8; ++r) {
      const int qrow = qt * 16 + mOff + r;
      const float val = acc[t][r] / lrun[r];
      const size_t row = (size_t)(b * Nt + qrow);
      if (col < 256) {
        Hmv[row * 2048 + h * 256 + col] = (__bf16)val;
        if ((col & 31) == 0) Hmv0[row * 64 + h * 8 + (col >> 5)] = (__bf16)val;
      } else {
        Hs[row * 128 + h * 16 + (col - 256)] = (__bf16)val;
      }
    }
  }
}

// ---------------------------------------------------------------- final combine
__global__ void combine_out(const float* __restrict__ base, const float* __restrict__ s2mv,
                            const float* __restrict__ s1, const float* __restrict__ s2,
                            float* __restrict__ out) {
  const long long MVN = (long long)ROWS * 512;
  const long long TOT = MVN + (long long)ROWS * 32;
  long long i = (long long)blockIdx.x * blockDim.x + threadIdx.x;
  if (i >= TOT) return;
  if (i < MVN) {
    const int j = (int)(i & 511);
    const long long row = i >> 9;
    float v = base[i];
    if ((j & 31) == 0) v += s2mv[row * 16 + (j >> 5)];
    out[i] = v;
  } else {
    const long long k = i - MVN;
    out[i] = s1[k] + s2[k];
  }
}

// ---------------------------------------------------------------- host launcher

extern "C" void kernel_launch(void* const* d_in, const int* in_sizes, int n_in,
                              void* d_out, int out_size, void* d_ws, size_t ws_size,
                              hipStream_t stream) {
  (void)in_sizes; (void)n_in; (void)out_size; (void)ws_size;
  const float* mv         = (const float*)d_in[0];
  const float* sc         = (const float*)d_in[1];
  const float* basis_q    = (const float*)d_in[2];
  const float* basis_k    = (const float*)d_in[3];
  const float* basis_pin  = (const float*)d_in[4];
  const float* w_q_mv     = (const float*)d_in[5];
  const float* w_k_mv     = (const float*)d_in[6];
  const float* w_v_mv     = (const float*)d_in[7];
  const float* w_q_s      = (const float*)d_in[8];
  const float* w_k_s      = (const float*)d_in[9];
  const float* w_v_s      = (const float*)d_in[10];
  const float* w_out_mv   = (const float*)d_in[11];
  const float* w_out_s    = (const float*)d_in[12];
  const float* w_out_s2mv = (const float*)d_in[13];
  const float* w_out_mv2s = (const float*)d_in[14];
  float* out = (float*)d_out;

  char* cur = (char*)d_ws;
  auto take = [&](size_t bytes) -> char* {
    char* p = cur;
    cur += (bytes + 255) & ~(size_t)255;
    return p;
  };
  __bf16* Xmv   = (__bf16*)take((size_t)ROWS * 512 * 2);
  __bf16* Xs    = (__bf16*)take((size_t)ROWS * 32 * 2);
  __bf16* WqE   = (__bf16*)take((size_t)2048 * 512 * 2);
  __bf16* WkE   = (__bf16*)take((size_t)256 * 512 * 2);
  __bf16* WvE   = (__bf16*)take((size_t)256 * 512 * 2);
  __bf16* WoE   = (__bf16*)take((size_t)512 * 2048 * 2);
  __bf16* Wqs   = (__bf16*)take((size_t)128 * 32 * 2);
  __bf16* Wks   = (__bf16*)take((size_t)16 * 32 * 2);
  __bf16* Wvs   = (__bf16*)take((size_t)16 * 32 * 2);
  __bf16* Wos   = (__bf16*)take((size_t)32 * 128 * 2);
  __bf16* Ws2mv = (__bf16*)take((size_t)16 * 128 * 2);
  __bf16* Wmv2s = (__bf16*)take((size_t)32 * 64 * 2);
  __bf16* Qmv   = (__bf16*)take((size_t)ROWS * 2048 * 2);
  __bf16* Kmv   = (__bf16*)take((size_t)ROWS * 256 * 2);
  __bf16* Vmv   = (__bf16*)take((size_t)ROWS * 256 * 2);
  __bf16* Qs    = (__bf16*)take((size_t)ROWS * 128 * 2);
  __bf16* Ks    = (__bf16*)take((size_t)ROWS * 16 * 2);
  __bf16* Vs    = (__bf16*)take((size_t)ROWS * 16 * 2);
  __bf16* Qfb   = (__bf16*)take((size_t)ROWS * Ht * FEAT * 2);
  __bf16* Kfb   = (__bf16*)take((size_t)ROWS * FEAT * 2);
  __bf16* Vtt   = (__bf16*)take((size_t)ROWS * VDIM * 2);
  __bf16* Hmv   = (__bf16*)take((size_t)ROWS * 2048 * 2);
  __bf16* Hs    = (__bf16*)take((size_t)ROWS * 128 * 2);
  __bf16* Hmv0  = (__bf16*)take((size_t)ROWS * 64 * 2);
  float* OutBase = (float*)take((size_t)ROWS * 512 * 4);
  float* S2MV    = (float*)take((size_t)ROWS * 16 * 4);
  float* OutS1   = (float*)take((size_t)ROWS * 32 * 4);
  float* OutS2   = (float*)take((size_t)ROWS * 32 * 4);

  auto cvt = [&](const float* s, __bf16* d, long long n) {
    cvt_bf16<<<(int)((n + 255) / 256), 256, 0, stream>>>(s, d, n);
  };
  cvt(mv, Xmv, (long long)ROWS * 512);
  cvt(sc, Xs, (long long)ROWS * 32);
  cvt(w_q_s, Wqs, 128 * 32);
  cvt(w_k_s, Wks, 16 * 32);
  cvt(w_v_s, Wvs, 16 * 32);
  cvt(w_out_s, Wos, 32 * 128);
  cvt(w_out_s2mv, Ws2mv, 16 * 128);
  cvt(w_out_mv2s, Wmv2s, 32 * 64);

  auto weff = [&](const float* w, __bf16* dst, int Oc, int Ic) {
    long long tot = (long long)Oc * 32 * Ic * 32;
    build_weff<<<(int)((tot + 255) / 256), 256, 0, stream>>>(w, basis_pin, dst, Oc, Ic);
  };
  weff(w_q_mv, WqE, 64, 16);
  weff(w_k_mv, WkE, 8, 16);
  weff(w_v_mv, WvE, 8, 16);
  weff(w_out_mv, WoE, 16, 64);

  // QKV projections (bf16 out)
  launch_gemm<__bf16, 4>(Xmv, WqE, Qmv, ROWS, 2048, 512, stream);
  launch_gemm<__bf16, 4>(Xmv, WkE, Kmv, ROWS, 256, 512, stream);
  launch_gemm<__bf16, 4>(Xmv, WvE, Vmv, ROWS, 256, 512, stream);
  launch_gemm<__bf16, 4>(Xs, Wqs, Qs, ROWS, 128, 32, stream);
  launch_gemm<__bf16, 1>(Xs, Wks, Ks, ROWS, 16, 32, stream);
  launch_gemm<__bf16, 1>(Xs, Wvs, Vs, ROWS, 16, 32, stream);

  const float scale = 0.125f;  // 1/sqrt(8*6 + 16)
  qf_pack<<<Bt * Ht * Nt, 32, 0, stream>>>(Qmv, Qs, basis_q, Qfb, scale);
  kf_pack<<<ROWS, 32, 0, stream>>>(Kmv, Ks, basis_k, Kfb);
  {
    long long tot = (long long)Bt * VDIM * Nt;
    vt_pack<<<(int)((tot + 255) / 256), 256, 0, stream>>>(Vmv, Vs, Vtt);
  }

  attn_fused<<<Bt * Ht * (Nt / 16), 32, 0, stream>>>(Qfb, Kfb, Vtt, Hmv, Hs, Hmv0);

  // Output projections (f32 out)
  launch_gemm<float, 4>(Hmv, WoE, OutBase, ROWS, 512, 2048, stream);
  launch_gemm<float, 1>(Hs, Ws2mv, S2MV, ROWS, 16, 128, stream);
  launch_gemm<float, 2>(Hs, Wos, OutS1, ROWS, 32, 128, stream);
  launch_gemm<float, 2>(Hmv0, Wmv2s, OutS2, ROWS, 32, 64, stream);

  const long long tot = (long long)ROWS * 512 + (long long)ROWS * 32;
  combine_out<<<(int)((tot + 255) / 256), 256, 0, stream>>>(OutBase, S2MV, OutS1, OutS2, out);
}
